// AutoregressiveRegression_76355928588986
// MI455X (gfx1250) — matrix operand order repaired and checked by reading, not compiled, not run on hardware
//
#include <hip/hip_runtime.h>
#include <hip/hip_bf16.h>

typedef __attribute__((ext_vector_type(16))) __bf16 v16bf;
typedef __attribute__((ext_vector_type(8)))  float  v8f;

#define UNITS   1024
#define BATCH   512
#define NCOLS   4096   // 4 * UNITS
#define KTILES  32     // UNITS / 32
#define OUTSTEPS 32

// ---------------------------------------------------------------------------
// Pack rec_kernel f32 [K=1024][N=4096] into bf16 WMMA-B fragment layout:
// tile (kb: 32 K, nb: 16 N); inside a tile each lane's 16 bf16 are contiguous
// (32 B / lane, 1 KB / tile). Lane mapping (contiguous k order, same as the A repack): lanes 0-15 hold
// K=0..15 (2 per dword slot), lanes 16-31 hold K=16..31; N = lane & 15.
// ---------------------------------------------------------------------------
__global__ __launch_bounds__(256)
void pack_rec_weights(const float* __restrict__ rec, __bf16* __restrict__ Wp) {
    size_t tid = (size_t)blockIdx.x * 256 + threadIdx.x;   // over 1024*4096
    int k = (int)(tid >> 12);
    int n = (int)(tid & 4095);
    int nb = n >> 4, n_in = n & 15;
    int kb = k >> 5, k_in = k & 31;
    int lane = n_in + ((k_in >> 4) << 4);
    int kl   = k_in & 15;
    size_t dst = ((size_t)(nb * KTILES + kb) << 9) + lane * 16 + (kl >> 1) * 2 + (kl & 1);
    Wp[dst] = (__bf16)rec[tid];
}

__global__ __launch_bounds__(256)
void zero_u4(uint4* __restrict__ p, int n) {
    int i = blockIdx.x * 256 + threadIdx.x;
    if (i < n) { uint4 z; z.x = z.y = z.z = z.w = 0u; p[i] = z; }
}

// ---------------------------------------------------------------------------
// Fused LSTM cell step:  z = hA @ Wp (+ x@kernel + bias) ; gates ; c,h update.
// Block: 256 threads = 8 waves. blockIdx.x = M-tile (16 rows of batch),
// blockIdx.y * 128 + wave*16 = unit-column base. Each wave owns one 16x16
// h-tile and 4 gate accumulators (4 x v8f).
// A operand (h, bf16, fragment layout) staged in LDS (32 KB slab per M-tile);
// B operand streamed from the L2-resident packed weights.
// Inner loop is branch-free: 2 ds_load_b128 (A) + 8 global_load_b128 (B, all
// four gates) + 4 in-place v_wmma_f32_16x16x32_bf16.
// ---------------------------------------------------------------------------
__global__ __launch_bounds__(256)
void lstm_cell_wmma(const float* __restrict__ xptr, int xstride,
                    const __bf16* __restrict__ hAr,   // A fragments, step t
                    __bf16* __restrict__ hAw,         // A fragments, step t+1
                    const __bf16* __restrict__ Wp,    // packed rec weights
                    const float* __restrict__ kern,   // [4][4096]
                    const float* __restrict__ bias,   // [4096]
                    float* __restrict__ c,            // [512][1024]
                    float* __restrict__ hf)           // [512][1024] f32 h
{
    __shared__ uint4 smemA[2048];                     // 32 KB A slab
    const int tid  = threadIdx.x;
    const int mb   = blockIdx.x;                      // 0..31
    const int wave = tid >> 5;
    const int lane = tid & 31;
    const int n_base = blockIdx.y * 128 + wave * 16;  // unit column base

    // cooperative load of this M-tile's A slab (16 x 1024 bf16, frag layout)
    const uint4* asrc = (const uint4*)hAr + (size_t)mb * 2048;
    #pragma unroll
    for (int i = 0; i < 8; ++i) smemA[i * 256 + tid] = asrc[i * 256 + tid];
    __syncthreads();

    v8f zv = {0.f, 0.f, 0.f, 0.f, 0.f, 0.f, 0.f, 0.f};
    v8f acc[4];
    #pragma unroll
    for (int g = 0; g < 4; ++g) acc[g] = zv;

    const v16bf* aFr = (const v16bf*)smemA;           // 32 frags per k-tile
    const int nbg0 = n_base >> 4;                     // N-tile id within gate 0
    // per-lane base into packed weights for gate 0, this wave's N tile
    const __bf16* wbase = Wp + (((size_t)nbg0 * KTILES) << 9) + (lane << 4);
    const size_t gstride = (size_t)64 * KTILES * 512; // gate stride (bf16 elems)

    #pragma unroll 2
    for (int kb = 0; kb < KTILES; ++kb) {
        v16bf a = aFr[kb * 32 + lane];
        const __bf16* wk = wbase + ((size_t)kb << 9);
        v16bf b0 = *(const v16bf*)(wk);
        v16bf b1 = *(const v16bf*)(wk + gstride);
        v16bf b2 = *(const v16bf*)(wk + 2 * gstride);
        v16bf b3 = *(const v16bf*)(wk + 3 * gstride);
        __builtin_prefetch(wk + 512, 0, 0);           // next k-tile, gate 0
        acc[0] = __builtin_amdgcn_wmma_f32_16x16x32_bf16(false, a, false, b0, (short)0, acc[0], false, false);
        acc[1] = __builtin_amdgcn_wmma_f32_16x16x32_bf16(false, a, false, b1, (short)0, acc[1], false, false);
        acc[2] = __builtin_amdgcn_wmma_f32_16x16x32_bf16(false, a, false, b2, (short)0, acc[2], false, false);
        acc[3] = __builtin_amdgcn_wmma_f32_16x16x32_bf16(false, a, false, b3, (short)0, acc[3], false, false);
    }

    // ---- epilogue: + x@kernel + bias, gates, c/h update, repack h ----
    const int n = n_base + (lane & 15);
    float kv[4][4], bb[4];
    #pragma unroll
    for (int g = 0; g < 4; ++g) {
        bb[g] = bias[g * UNITS + n];
        #pragma unroll
        for (int f = 0; f < 4; ++f) kv[g][f] = kern[f * NCOLS + g * UNITS + n];
    }
    const int mrow0 = mb * 16 + ((lane >> 4) << 3);   // +8 for upper half-lanes

    #pragma unroll
    for (int r = 0; r < 8; ++r) {
        const int m = mrow0 + r;
        const float* xr = xptr + (size_t)m * xstride;
        float x0 = xr[0], x1 = xr[1], x2 = xr[2], x3 = xr[3];
        float zi = acc[0][r] + bb[0] + x0*kv[0][0] + x1*kv[0][1] + x2*kv[0][2] + x3*kv[0][3];
        float zf = acc[1][r] + bb[1] + x0*kv[1][0] + x1*kv[1][1] + x2*kv[1][2] + x3*kv[1][3];
        float zg = acc[2][r] + bb[2] + x0*kv[2][0] + x1*kv[2][1] + x2*kv[2][2] + x3*kv[2][3];
        float zo = acc[3][r] + bb[3] + x0*kv[3][0] + x1*kv[3][1] + x2*kv[3][2] + x3*kv[3][3];
        float ig = 1.f / (1.f + __expf(-zi));
        float fg = 1.f / (1.f + __expf(-zf));
        float og = 1.f / (1.f + __expf(-zo));
        float gg = tanhf(zg);
        size_t idx = (size_t)m * UNITS + n;
        float cn = fg * c[idx] + ig * gg;
        c[idx] = cn;
        float hn = og * tanhf(cn);
        hf[idx] = hn;
        // repack h into A-fragment layout (k = unit index n) for next step
        int m_in = m & 15;
        int kb2  = n >> 5, k_in = n & 31;
        int w16  = k_in & 15;
        int lane2 = m_in + ((k_in >> 4) << 4);
        int kk    = w16;
        int vg    = kk >> 1;
        hAw[((size_t)(mb * KTILES + kb2) << 9) + lane2 * 16 + vg * 2 + (kk & 1)] = (__bf16)hn;
    }
}

// ---------------------------------------------------------------------------
// Dense head: pred[b][j] = h[b] . dense_w[:,j] + db[j]  (j < 4)
// Also writes the [B, 32, 4] output slice for step `sstep`.
// ---------------------------------------------------------------------------
__global__ __launch_bounds__(256)
void dense_head(const float* __restrict__ hf, const float* __restrict__ dw,
                const float* __restrict__ db, float* __restrict__ pred,
                float* __restrict__ out, int sstep)
{
    int t = blockIdx.x * 256 + threadIdx.x;
    if (t >= BATCH * 4) return;
    int b = t >> 2, j = t & 3;
    const float* hr = hf + (size_t)b * UNITS;
    float s = db[j];
    for (int u = 0; u < UNITS; ++u) s += hr[u] * dw[u * 4 + j];
    pred[t] = s;
    out[((size_t)b * OUTSTEPS + sstep) * 4 + j] = s;
}

extern "C" void kernel_launch(void* const* d_in, const int* in_sizes, int n_in,
                              void* d_out, int out_size, void* d_ws, size_t ws_size,
                              hipStream_t stream) {
    const float* x    = (const float*)d_in[0];  // [512,128,4]
    const float* kern = (const float*)d_in[1];  // [4,4096]
    const float* rec  = (const float*)d_in[2];  // [1024,4096]
    const float* bias = (const float*)d_in[3];  // [4096]
    const float* dw   = (const float*)d_in[4];  // [1024,4]
    const float* db   = (const float*)d_in[5];  // [4]
    float* out = (float*)d_out;                 // [512,32,4]

    char* ws = (char*)d_ws;
    __bf16* Wp  = (__bf16*)(ws);                          // 8 MB packed weights
    __bf16* hA0 = (__bf16*)(ws +  8u * 1024 * 1024);      // 1 MB A frags (ping)
    __bf16* hA1 = (__bf16*)(ws +  9u * 1024 * 1024);      // 1 MB A frags (pong)
    float*  hf  = (float*) (ws + 10u * 1024 * 1024);      // 2 MB h (f32)
    float*  c   = (float*) (ws + 12u * 1024 * 1024);      // 2 MB c
    float*  pr  = (float*) (ws + 14u * 1024 * 1024);      // 8 KB pred feedback

    // one-time (per call) weight packing; L2-resident thereafter
    pack_rec_weights<<<(UNITS * NCOLS) / 256, 256, 0, stream>>>(rec, Wp);
    // zero initial state: c (131072 uint4) and hA0 (65536 uint4)
    zero_u4<<<131072 / 256, 256, 0, stream>>>((uint4*)c, 131072);
    zero_u4<<< 65536 / 256, 256, 0, stream>>>((uint4*)hA0, 65536);

    __bf16* hA[2] = {hA0, hA1};
    dim3 cgrid(32, 8);

    // warmup scan over T=128 input steps
    for (int t = 0; t < 128; ++t)
        lstm_cell_wmma<<<cgrid, 256, 0, stream>>>(x + t * 4, 512,
                                                  hA[t & 1], hA[(t + 1) & 1],
                                                  Wp, kern, bias, c, hf);
    dense_head<<<8, 256, 0, stream>>>(hf, dw, db, pr, out, 0);

    // autoregressive decode: feed prediction back as the 4-feature input
    for (int s = 1; s < OUTSTEPS; ++s) {
        int tt = 127 + s;
        lstm_cell_wmma<<<cgrid, 256, 0, stream>>>(pr, 4,
                                                  hA[tt & 1], hA[(tt + 1) & 1],
                                                  Wp, kern, bias, c, hf);
        dense_head<<<8, 256, 0, stream>>>(hf, dw, db, pr, out, s);
    }
}
